// PragnosiaTransformer_26551487824579
// MI455X (gfx1250) — compile-verified, matrix operand order checked by reading, not run on hardware
//
#include <hip/hip_runtime.h>
#include <math.h>

// ---------------- problem constants ----------------
constexpr int kB = 2, kS = 1024, kH = 1024;
constexpr int kNH = 16, kKVH = 4, kHD = 64;
constexpr int kE = 8, kI = 2816, kL = 2;
constexpr int kT = kB * kS;              // 2048 tokens
constexpr float kEps = 1e-6f;
constexpr float kTheta = 10000.f;

// ---------------- WMMA types / helpers ----------------
typedef __attribute__((ext_vector_type(16))) __bf16 v16bf;
typedef __attribute__((ext_vector_type(8)))  __bf16 v8bf;
typedef __attribute__((ext_vector_type(4)))  __bf16 v4bf;
typedef __attribute__((ext_vector_type(8)))  float  v8f;

__device__ __forceinline__ __bf16 f2bf(float f) { return static_cast<__bf16>(f); }

__device__ __forceinline__ v4bf f2bf4(float4 v) {
  v4bf r;
  r[0] = f2bf(v.x); r[1] = f2bf(v.y); r[2] = f2bf(v.z); r[3] = f2bf(v.w);
  return r;
}

__device__ __forceinline__ v8f wmma_bf16(v16bf a, v16bf b, v8f c) {
  // D(16x16,f32) = A(16x32,bf16) x B(32x16,bf16) + C
  return __builtin_amdgcn_wmma_f32_16x16x32_bf16(false, a, false, b, (short)0, c,
                                                 false, false);
}

// Load a 16-element bf16 fragment: two contiguous 16-byte chunks per lane.
// A/B operand K-mapping: elems 0..7 -> K = half*8 + j ; elems 8..15 -> K = 16 + half*8 + (j-8)
__device__ __forceinline__ v16bf lds_frag(const __bf16* rowBase, int half) {
  v8bf lo = *(const v8bf*)(rowBase + half * 8);
  v8bf hi = *(const v8bf*)(rowBase + 16 + half * 8);
  return __builtin_shufflevector(lo, hi, 0,1,2,3,4,5,6,7,8,9,10,11,12,13,14,15);
}

// Build a fragment from fp32 memory (global or LDS): two contiguous 8-float runs.
__device__ __forceinline__ v16bf frag_from_f32(const float* base, int half) {
  float4 a = *(const float4*)(base + half*8);
  float4 b = *(const float4*)(base + half*8 + 4);
  float4 c = *(const float4*)(base + 16 + half*8);
  float4 d = *(const float4*)(base + 16 + half*8 + 4);
  v16bf r;
  r[0]=f2bf(a.x);  r[1]=f2bf(a.y);  r[2]=f2bf(a.z);  r[3]=f2bf(a.w);
  r[4]=f2bf(b.x);  r[5]=f2bf(b.y);  r[6]=f2bf(b.z);  r[7]=f2bf(b.w);
  r[8]=f2bf(c.x);  r[9]=f2bf(c.y);  r[10]=f2bf(c.z); r[11]=f2bf(c.w);
  r[12]=f2bf(d.x); r[13]=f2bf(d.y); r[14]=f2bf(d.z); r[15]=f2bf(d.w);
  return r;
}

// ---------------- generic tiled GEMM (bf16 WMMA, f32 accum) ----------------
// C[M,N] = A[M,K] * B[K,N]  (fp32 in memory, bf16 in the MACs)
// MODE 0: C = AB ; MODE 1: C = AB + addend ; MODE 2: C += rowscale[m*rsStride] * AB
// Block tile 128x128, BK=32, 8 waves (2x4), wave tile 64x32 = 4x2 WMMA tiles.
// Double-buffered: tile k+1 global loads are in flight during tile-k WMMAs.
constexpr int LDT = 40;   // LDS row stride in bf16 elems (80B: keeps 16B alignment)

template <int MODE>
__global__ __launch_bounds__(256) void gemm_bf16(
    const float* __restrict__ A, const float* __restrict__ Bm, float* __restrict__ C,
    const float* __restrict__ addend, const float* __restrict__ rowscale, int rsStride,
    int M, int N, int K)
{
  __shared__ __align__(16) __bf16 As[128 * LDT];  // 128(M) x 32(K)
  __shared__ __align__(16) __bf16 Bs[128 * LDT];  // 128(N) x 32(K)  (B^T tile)
  const int tid  = threadIdx.x;
  const int lane = tid & 31, wave = tid >> 5;
  const int ln   = lane & 15, half = lane >> 4;
  const int wm   = wave >> 2, wn = wave & 3;       // 2x4 waves
  const int blockM = blockIdx.y * 128, blockN = blockIdx.x * 128;

  const int nk = K >> 5;
  // A staging: idx -> row = idx>>3 (0..127), c4 = (idx&7)*4 (K-contiguous float4)
  // B staging: idx -> n = idx&127 (lane-consecutive, coalesced), kg = idx>>7 (0..7),
  //            4 K-consecutive values at fixed n -> one packed b64 LDS store.
  float4 aReg[4], bReg[4];
  #pragma unroll
  for (int it = 0; it < 4; ++it) {
    int idx = tid + it * 256;
    aReg[it] = *(const float4*)(A + (size_t)(blockM + (idx >> 3)) * K + ((idx & 7) * 4));
    const float* bp = Bm + (size_t)((idx >> 7) * 4) * N + blockN + (idx & 127);
    bReg[it] = make_float4(bp[0], bp[(size_t)N], bp[(size_t)2*N], bp[(size_t)3*N]);
  }

  v8f acc[4][2] = {};
  for (int kt = 0; kt < nk; ++kt) {
    if (kt) __syncthreads();                       // prev tile's LDS reads done
    // stage current tile (convert f32 -> bf16, packed 8B stores)
    #pragma unroll
    for (int it = 0; it < 4; ++it) {
      int idx = tid + it * 256;
      *(v4bf*)&As[(idx >> 3) * LDT + ((idx & 7) * 4)] = f2bf4(aReg[it]);
      *(v4bf*)&Bs[(idx & 127) * LDT + ((idx >> 7) * 4)] = f2bf4(bReg[it]);
    }
    __syncthreads();
    // issue next tile's global loads (overlap with WMMA below)
    if (kt + 1 < nk) {
      const int k0 = (kt + 1) * 32;
      #pragma unroll
      for (int it = 0; it < 4; ++it) {
        int idx = tid + it * 256;
        aReg[it] = *(const float4*)(A + (size_t)(blockM + (idx >> 3)) * K + k0 + ((idx & 7) * 4));
        const float* bp = Bm + (size_t)(k0 + (idx >> 7) * 4) * N + blockN + (idx & 127);
        bReg[it] = make_float4(bp[0], bp[(size_t)N], bp[(size_t)2*N], bp[(size_t)3*N]);
      }
      if (kt + 2 < nk) {                           // warm L2 two tiles ahead
        const int k2 = (kt + 2) * 32;
        __builtin_prefetch(A  + (size_t)(blockM + (tid >> 3)) * K + k2, 0, 1);
        __builtin_prefetch(Bm + (size_t)(k2 + (tid >> 7) * 4) * N + blockN + (tid & 127), 0, 1);
      }
    }
    // fragments + 8 WMMAs
    v16bf a[4], b[2];
    #pragma unroll
    for (int tm = 0; tm < 4; ++tm)
      a[tm] = lds_frag(&As[(wm*64 + tm*16 + ln) * LDT], half);
    #pragma unroll
    for (int tn = 0; tn < 2; ++tn)
      b[tn] = lds_frag(&Bs[(wn*32 + tn*16 + ln) * LDT], half);
    #pragma unroll
    for (int tm = 0; tm < 4; ++tm)
      #pragma unroll
      for (int tn = 0; tn < 2; ++tn)
        acc[tm][tn] = wmma_bf16(a[tm], b[tn], acc[tm][tn]);
  }

  // epilogue: C/D layout -> lane n = ln, row = vgpr_j + 8*half
  #pragma unroll
  for (int tm = 0; tm < 4; ++tm) {
    #pragma unroll
    for (int tn = 0; tn < 2; ++tn) {
      const int col  = blockN + wn*32 + tn*16 + ln;
      const int row0 = blockM + wm*64 + tm*16 + half*8;
      size_t idx = (size_t)row0 * N + col;
      if (MODE == 0) {
        #pragma unroll
        for (int j = 0; j < 8; ++j) { C[idx] = acc[tm][tn][j]; idx += N; }
      } else if (MODE == 1) {
        #pragma unroll
        for (int j = 0; j < 8; ++j) { C[idx] = acc[tm][tn][j] + addend[idx]; idx += N; }
      } else {
        const float* rs = rowscale + (size_t)row0 * rsStride;
        #pragma unroll
        for (int j = 0; j < 8; ++j) { C[idx] += rs[(size_t)j * rsStride] * acc[tm][tn][j]; idx += N; }
      }
    }
  }
}

// ---------------- flash attention (WMMA), 1 wave per (b, head, 16-q tile) ----------------
__global__ __launch_bounds__(32) void attn_kernel(
    const float* __restrict__ q, const float* __restrict__ kmat,
    const float* __restrict__ vmat, float* __restrict__ o)
{
  __shared__ __align__(16) float sS[16 * 32];
  __shared__ __align__(16) float sP[16 * 32];
  __shared__ float sAlpha[16], sSum[16];
  const int lane = threadIdx.x, ln = lane & 15, half = lane >> 4;
  int bid = blockIdx.x;
  const int qt = bid & 63;  bid >>= 6;
  const int hh = bid & 15;  bid >>= 4;
  const int b  = bid;
  const int kvh = hh / (kNH / kKVH);
  const float scale = 0.125f;                      // 1/sqrt(64)

  // Q fragment (A operand): row = qt*16 + ln; two HD chunks of 32
  const int qrow = qt * 16 + ln;
  const float* qp = q + (((size_t)(b*kS) + qrow)*kNH + hh)*kHD;
  v16bf aq[2];
  aq[0] = frag_from_f32(qp, half);
  aq[1] = frag_from_f32(qp + 32, half);

  v8f co[4] = {};
  float rowMax = -3.0e38f, rowSum = 0.f;
  const int ktMax = (qt*16 + 15) >> 5;
  for (int kt = 0; kt <= ktMax; ++kt) {
    // scores S(16x32) = Q(16x64) * K^T  -> two 16-key subtiles
    #pragma unroll
    for (int ns = 0; ns < 2; ++ns) {
      int key = kt*32 + ns*16 + ln;
      const float* kp = kmat + (((size_t)(b*kS) + key)*kKVH + kvh)*kHD;
      v16bf bk0 = frag_from_f32(kp, half);
      v16bf bk1 = frag_from_f32(kp + 32, half);
      v8f cs = {};
      cs = wmma_bf16(aq[0], bk0, cs);
      cs = wmma_bf16(aq[1], bk1, cs);
      #pragma unroll
      for (int j = 0; j < 8; ++j) {
        int rloc = j + half*8;
        int qg = qt*16 + rloc, kg = kt*32 + ns*16 + ln;
        sS[rloc*32 + ns*16 + ln] = (kg <= qg) ? cs[j] * scale : -1.0e30f;
      }
    }
    __syncthreads();
    if (lane < 16) {                               // lane r owns row r statistics
      const int r = lane;
      float mx = rowMax;
      #pragma unroll
      for (int j = 0; j < 32; ++j) mx = fmaxf(mx, sS[r*32 + j]);
      float alpha = __expf(rowMax - mx);
      float sum = rowSum * alpha;
      #pragma unroll
      for (int j = 0; j < 32; ++j) {
        float pp = __expf(sS[r*32 + j] - mx);
        sP[r*32 + j] = pp;
        sum += pp;
      }
      rowMax = mx; rowSum = sum;
      sAlpha[r] = alpha; sSum[r] = sum;
    }
    __syncthreads();
    #pragma unroll
    for (int nt = 0; nt < 4; ++nt)
      #pragma unroll
      for (int j = 0; j < 8; ++j)
        co[nt][j] *= sAlpha[j + half*8];
    // P fragment (A operand): row = ln, K = 32 keys (contiguous in sP)
    v16bf ap = frag_from_f32(&sP[ln*32], half);
    // O(16x64) += P(16x32) * V(32x64) -> four 16-col subtiles
    #pragma unroll
    for (int nt = 0; nt < 4; ++nt) {
      v16bf bv;
      #pragma unroll
      for (int j = 0; j < 16; ++j) {
        int off = (j < 8) ? (half*8 + j) : (16 + half*8 + (j - 8));
        int key = kt*32 + off;
        bv[j] = f2bf(vmat[(((size_t)(b*kS) + key)*kKVH + kvh)*kHD + nt*16 + ln]);
      }
      co[nt] = wmma_bf16(ap, bv, co[nt]);
    }
    __syncthreads();
  }
  #pragma unroll
  for (int nt = 0; nt < 4; ++nt)
    #pragma unroll
    for (int j = 0; j < 8; ++j) {
      int rloc = j + half*8;
      int qg = qt*16 + rloc;
      o[(((size_t)(b*kS) + qg)*kNH + hh)*kHD + nt*16 + ln] = co[nt][j] / sSum[rloc];
    }
}

// ---------------- small helper kernels ----------------
__global__ void embed_kernel(const int* __restrict__ ids, const float* __restrict__ emb,
                             float* __restrict__ x) {
  int t = blockIdx.x;
  int row = ids[t];
  for (int i = threadIdx.x; i < kH; i += blockDim.x)
    x[(size_t)t*kH + i] = emb[(size_t)row*kH + i];
}

__global__ __launch_bounds__(256) void rmsnorm_kernel(const float* __restrict__ x,
    const float* __restrict__ w, float* __restrict__ y)
{
  __shared__ float red[256];
  const int t = blockIdx.x, tid = threadIdx.x;
  const float* xr = x + (size_t)t * kH;
  float s = 0.f;
  for (int i = tid; i < kH; i += 256) { float v = xr[i]; s += v * v; }
  red[tid] = s; __syncthreads();
  for (int st = 128; st > 0; st >>= 1) {
    if (tid < st) red[tid] += red[tid + st];
    __syncthreads();
  }
  float inv = rsqrtf(red[0] / (float)kH + kEps);
  for (int i = tid; i < kH; i += 256)
    y[(size_t)t*kH + i] = xr[i] * w[i] * inv;
}

__global__ __launch_bounds__(32) void rope_kernel(float* __restrict__ buf, int nheads) {
  int bidx = blockIdx.x;
  int head = bidx % nheads;
  int t = bidx / nheads;
  int s = t % kS;
  int d = threadIdx.x;                             // 0..31 -> pair (d, d+32)
  float inv_freq = powf(kTheta, -(float)(2*d) / (float)kHD);
  float ang = (float)s * inv_freq;
  float c = cosf(ang), sn = sinf(ang);
  float* p = buf + ((size_t)t * nheads + head) * kHD;
  float x0 = p[d], x1 = p[d + 32];
  p[d]      = x0 * c - x1 * sn;
  p[d + 32] = x1 * c + x0 * sn;
}

__global__ void router_kernel(const float* __restrict__ h, const float* __restrict__ rw,
                              float* __restrict__ we)
{
  int t = blockIdx.x * blockDim.x + threadIdx.x;
  if (t >= kT) return;
  const float* hr = h + (size_t)t * kH;
  float lg[kE];
  #pragma unroll
  for (int e = 0; e < kE; ++e) lg[e] = 0.f;
  for (int i = 0; i < kH; ++i) {
    float hv = hr[i];
    #pragma unroll
    for (int e = 0; e < kE; ++e) lg[e] += hv * rw[(size_t)i*kE + e];
  }
  float mx = lg[0];
  #pragma unroll
  for (int e = 1; e < kE; ++e) mx = fmaxf(mx, lg[e]);
  float p[kE], sum = 0.f;
  #pragma unroll
  for (int e = 0; e < kE; ++e) { p[e] = __expf(lg[e] - mx); sum += p[e]; }
  #pragma unroll
  for (int e = 0; e < kE; ++e) p[e] /= sum;
  int i1 = 0;
  #pragma unroll
  for (int e = 1; e < kE; ++e) if (p[e] > p[i1]) i1 = e;
  int i2 = (i1 == 0) ? 1 : 0;
  #pragma unroll
  for (int e = 0; e < kE; ++e) if (e != i1 && p[e] > p[i2]) i2 = e;
  float wsum = p[i1] + p[i2];
  #pragma unroll
  for (int e = 0; e < kE; ++e)
    we[(size_t)t*kE + e] = (e == i1) ? p[i1]/wsum : (e == i2) ? p[i2]/wsum : 0.f;
}

__global__ void silu_mul_kernel(float* __restrict__ g, const float* __restrict__ u, int n) {
  int i = blockIdx.x * blockDim.x + threadIdx.x;
  if (i < n) { float x = g[i]; g[i] = (x / (1.f + __expf(-x))) * u[i]; }
}

__global__ void zero_kernel(float* __restrict__ p, int n) {
  int i = blockIdx.x * blockDim.x + threadIdx.x;
  if (i < n) p[i] = 0.f;
}

__global__ void add_kernel(float* __restrict__ x, const float* __restrict__ y, int n) {
  int i = blockIdx.x * blockDim.x + threadIdx.x;
  if (i < n) x[i] += y[i];
}

// ---------------- orchestration ----------------
extern "C" void kernel_launch(void* const* d_in, const int* in_sizes, int n_in,
                              void* d_out, int out_size, void* d_ws, size_t ws_size,
                              hipStream_t stream)
{
  const int*   ids = (const int*)  d_in[0];
  const float* emb = (const float*)d_in[1];
  const float* ln1 = (const float*)d_in[2];
  const float* qw  = (const float*)d_in[3];
  const float* kw  = (const float*)d_in[4];
  const float* vw  = (const float*)d_in[5];
  const float* ow  = (const float*)d_in[6];
  const float* ln2 = (const float*)d_in[7];
  const float* rtw = (const float*)d_in[8];
  const float* wg  = (const float*)d_in[9];
  const float* wu  = (const float*)d_in[10];
  const float* wd  = (const float*)d_in[11];
  const float* fnw = (const float*)d_in[12];
  float* out = (float*)d_out;

  float* p = (float*)d_ws;
  float* x    = p; p += (size_t)kT * kH;
  float* h    = p; p += (size_t)kT * kH;
  float* qb   = p; p += (size_t)kT * kNH  * kHD;
  float* kb   = p; p += (size_t)kT * kKVH * kHD;
  float* vb   = p; p += (size_t)kT * kKVH * kHD;
  float* ao   = p; p += (size_t)kT * kNH  * kHD;
  float* gate = p; p += (size_t)kT * kI;
  float* up   = p; p += (size_t)kT * kI;
  float* moe  = p; p += (size_t)kT * kH;
  float* we   = p; p += (size_t)kT * kE;

  const int QO  = kNH  * kHD;   // 1024
  const int KVO = kKVH * kHD;   // 256

  embed_kernel<<<kT, 256, 0, stream>>>(ids, emb, x);

  for (int l = 0; l < kL; ++l) {
    rmsnorm_kernel<<<kT, 256, 0, stream>>>(x, ln1 + (size_t)l*kH, h);
    gemm_bf16<0><<<dim3(QO/128,  kT/128), 256, 0, stream>>>(h, qw + (size_t)l*kH*QO,  qb,
        nullptr, nullptr, 0, kT, QO,  kH);
    gemm_bf16<0><<<dim3(KVO/128, kT/128), 256, 0, stream>>>(h, kw + (size_t)l*kH*KVO, kb,
        nullptr, nullptr, 0, kT, KVO, kH);
    gemm_bf16<0><<<dim3(KVO/128, kT/128), 256, 0, stream>>>(h, vw + (size_t)l*kH*KVO, vb,
        nullptr, nullptr, 0, kT, KVO, kH);
    rope_kernel<<<kT*kNH,  32, 0, stream>>>(qb, kNH);
    rope_kernel<<<kT*kKVH, 32, 0, stream>>>(kb, kKVH);
    attn_kernel<<<kB*kNH*(kS/16), 32, 0, stream>>>(qb, kb, vb, ao);
    // x = x + attn_out @ o_w
    gemm_bf16<1><<<dim3(kH/128, kT/128), 256, 0, stream>>>(ao, ow + (size_t)l*QO*kH, x,
        x, nullptr, 0, kT, kH, QO);

    rmsnorm_kernel<<<kT, 256, 0, stream>>>(x, ln2 + (size_t)l*kH, h);
    router_kernel<<<(kT + 255)/256, 256, 0, stream>>>(h, rtw + (size_t)l*kH*kE, we);
    zero_kernel<<<(kT*kH + 255)/256, 256, 0, stream>>>(moe, kT*kH);
    for (int e = 0; e < kE; ++e) {
      const float* wge = wg + ((size_t)l*kE + e) * (size_t)kH * kI;
      const float* wue = wu + ((size_t)l*kE + e) * (size_t)kH * kI;
      const float* wde = wd + ((size_t)l*kE + e) * (size_t)kI * kH;
      gemm_bf16<0><<<dim3(kI/128, kT/128), 256, 0, stream>>>(h, wge, gate,
          nullptr, nullptr, 0, kT, kI, kH);
      gemm_bf16<0><<<dim3(kI/128, kT/128), 256, 0, stream>>>(h, wue, up,
          nullptr, nullptr, 0, kT, kI, kH);
      silu_mul_kernel<<<((kT*kI) + 255)/256, 256, 0, stream>>>(gate, up, kT*kI);
      // moe += we[:,e] * (silu(gate)*up) @ w_down[e]
      gemm_bf16<2><<<dim3(kH/128, kT/128), 256, 0, stream>>>(gate, wde, moe,
          nullptr, we + e, kE, kT, kH, kI);
    }
    add_kernel<<<(kT*kH + 255)/256, 256, 0, stream>>>(x, moe, kT*kH);
  }
  rmsnorm_kernel<<<kT, 256, 0, stream>>>(x, fnw, out);

  (void)in_sizes; (void)n_in; (void)out_size; (void)ws_size;
}
